// Decoder_63977832841480
// MI455X (gfx1250) — compile-verified, matrix operand order checked by reading, not run on hardware
//
#include <hip/hip_runtime.h>
#include <cstdint>
#include <cstddef>

// ---------------- problem constants (from reference) ----------------
#define VOC   32000
#define EMB   512
#define HID   1024
#define BATCH 32
#define MLEN  64
#define TSTEP 63            // ML-1 teacher-forced steps
#define G3    3072          // 3*H
#define SOS_TOK 1
#define NROWS (TSTEP*BATCH) // 2016

typedef __bf16 bf16;
typedef __attribute__((ext_vector_type(16))) __bf16 v16bf;
typedef __attribute__((ext_vector_type(8)))  __bf16 v8bf;
typedef __attribute__((ext_vector_type(8)))  float  v8f;
typedef __attribute__((ext_vector_type(4)))  int    v4i;

// -------- optional CDNA5 async global->LDS path (ASYNCcnt) --------
#if defined(__has_builtin)
#  if __has_builtin(__builtin_amdgcn_global_load_async_to_lds_b128) && \
      __has_builtin(__builtin_amdgcn_s_wait_asynccnt)
#    define USE_ASYNC_LDS 1
#  endif
#endif
#ifndef USE_ASYNC_LDS
#  define USE_ASYNC_LDS 0
#endif

#if USE_ASYNC_LDS
// Toolchain signature (from diagnostic): (int4 AS1*, int4 AS3*, imm, imm)
typedef __attribute__((address_space(1))) v4i* as1_v4i;
typedef __attribute__((address_space(3))) v4i* as3_v4i;
__device__ __forceinline__ void async_copy_b128(const bf16* g, bf16* l) {
  __builtin_amdgcn_global_load_async_to_lds_b128((as1_v4i)g, (as3_v4i)l, 0, 0);
}
#endif

// ---------------------------------------------------------------
// Fragment load (global) for 16-bit 16x32 (rows x K) operand, CDNA5 layout:
//   lane 0..15  : row = base+lane,    K = k+[0..7]  and k+[16..23]
//   lane 16..31 : row = base+lane-16, K = k+[8..15] and k+[24..31]
// Two 128-bit loads per lane.
// ---------------------------------------------------------------
__device__ __forceinline__ v16bf load_frag(const bf16* __restrict__ base, int ld,
                                           int rowBase, int kBase, int lane) {
  int r   = rowBase + (lane & 15);
  int off = (lane >> 4) * 8;
  const bf16* p = base + (size_t)r * ld + kBase + off;
  union { v16bf v; v8bf h[2]; } u;
  u.h[0] = *(const v8bf*)(p);
  u.h[1] = *(const v8bf*)(p + 16);
  return u.v;
}

#define WMMA_BF16(a, b, c) \
  __builtin_amdgcn_wmma_f32_16x16x32_bf16(false, (a), false, (b), (short)0, (c), false, false)

// ===============================================================
// gemm_small: C = A[M,K] @ W[N,K]^T + bias[N]   (register-only)
// One wave -> 16(M) x 64(N) tile, register double-buffered K loop.
// Used for the skinny sequential step GEMMs (M = 32).
// ===============================================================
__global__ void gemm_small(const bf16* __restrict__ A,
                           const bf16* __restrict__ W,
                           const float* __restrict__ bias,
                           float* __restrict__ C,
                           int M, int N, int K) {
  int wave = (blockIdx.x * blockDim.x + threadIdx.x) >> 5;
  int lane = threadIdx.x & 31;
  int ntiles = N >> 6;
  int mtiles = M >> 4;
  if (wave >= ntiles * mtiles) return;   // wave-uniform (EXEC stays all-1)
  int tn = wave % ntiles;
  int tm = wave / ntiles;
  int mBase = tm << 4;
  int nBase = tn << 6;

  v8f acc[4] = {v8f{}, v8f{}, v8f{}, v8f{}};

  // prologue: load chunk 0
  v16bf a  = load_frag(A, K, mBase,       0, lane);
  v16bf b0 = load_frag(W, K, nBase +  0,  0, lane);
  v16bf b1 = load_frag(W, K, nBase + 16,  0, lane);
  v16bf b2 = load_frag(W, K, nBase + 32,  0, lane);
  v16bf b3 = load_frag(W, K, nBase + 48,  0, lane);

  for (int k = 0; k < K; k += 32) {
    int kn = k + 32;
    v16bf an = a, c0 = b0, c1 = b1, c2 = b2, c3 = b3;
    if (kn < K) {          // prefetch next chunk while WMMAs execute
      an = load_frag(A, K, mBase,      kn, lane);
      c0 = load_frag(W, K, nBase +  0, kn, lane);
      c1 = load_frag(W, K, nBase + 16, kn, lane);
      c2 = load_frag(W, K, nBase + 32, kn, lane);
      c3 = load_frag(W, K, nBase + 48, kn, lane);
    }
    acc[0] = WMMA_BF16(a, b0, acc[0]);
    acc[1] = WMMA_BF16(a, b1, acc[1]);
    acc[2] = WMMA_BF16(a, b2, acc[2]);
    acc[3] = WMMA_BF16(a, b3, acc[3]);
    a = an; b0 = c0; b1 = c1; b2 = c2; b3 = c3;
  }

  int col  = lane & 15;
  int mOff = (lane >> 4) * 8;
#pragma unroll
  for (int j = 0; j < 4; ++j) {
    int n   = nBase + j * 16 + col;
    float b = bias[n];
#pragma unroll
    for (int i = 0; i < 8; ++i)
      C[(size_t)(mBase + mOff + i) * N + n] = acc[j][i] + b;
  }
}

// ===============================================================
// gemm_big: C = A[M,K] @ W[N,K]^T + bias[N]
// Block = 256 threads (8 waves). Block tile = 128(M) x 64(N).
// B chunk (64 rows x 32 K, bf16) is cooperatively staged into LDS,
// double-buffered, shared by all 8 waves (8x less L2 W-traffic).
// Staging uses GLOBAL_LOAD_ASYNC_TO_LDS_B128 (ASYNCcnt) when the
// builtin exists; otherwise global->reg->ds_store. A fragments are
// register double-buffered per wave. Padded LDS rows (40 halves =
// 80B stride) keep ds_load_b128 16B-aligned and conflict-light.
// Ragged M handled by row clamping + wave-uniform store guard.
// ===============================================================
#define BM 128
#define BN 64
#define LROW 40            // halves per LDS row (32 data + 8 pad)

__global__ void gemm_big(const bf16* __restrict__ A,
                         const bf16* __restrict__ W,
                         const float* __restrict__ bias,
                         float* __restrict__ C,
                         int M, int N, int K) {
  __shared__ bf16 Bsh[2][BN * LROW];

  int tid   = threadIdx.x;
  int wave  = tid >> 5;
  int lane  = tid & 31;
  int nBase = blockIdx.x * BN;
  int mBase = blockIdx.y * BM + wave * 16;

  // cooperative B load: one 16B (8-half) segment per thread, 64x32 total
  int brow = tid >> 2;               // 0..63
  int bcol = (tid & 3) * 8;          // 0,8,16,24 halves
  const bf16* gB = W + (size_t)(nBase + brow) * K + bcol;
  bf16* sB0 = &Bsh[0][brow * LROW + bcol];
  bf16* sB1 = &Bsh[1][brow * LROW + bcol];

  // per-wave A fragment addressing (rows clamped for ragged tail)
  int ar = mBase + (lane & 15);
  if (ar > M - 1) ar = M - 1;
  int aoff = (lane >> 4) * 8;
  const bf16* gA = A + (size_t)ar * K + aoff;

  v8f acc[4] = {v8f{}, v8f{}, v8f{}, v8f{}};

  int nchunks = K >> 5;
  int buf = 0;

#if USE_ASYNC_LDS
  // ---------------- async double-buffered pipeline ----------------
  async_copy_b128(gB, sB0);                    // chunk 0 -> LDS[0]
  v8bf aLo = *(const v8bf*)(gA);
  v8bf aHi = *(const v8bf*)(gA + 16);
  __builtin_amdgcn_s_wait_asynccnt(0);
  __syncthreads();

  for (int kc = 0; kc < nchunks; ++kc) {
    int kn = (kc + 1) << 5;
    if (kn < K)                                // async-fill other buffer
      async_copy_b128(gB + kn, buf ? sB0 : sB1);

    union { v16bf v; v8bf h[2]; } au;
    au.h[0] = aLo; au.h[1] = aHi;
    v16bf a = au.v;
    if (kn < K) {                              // prefetch next A frag (regs)
      aLo = *(const v8bf*)(gA + kn);
      aHi = *(const v8bf*)(gA + kn + 16);
    }

    const bf16* bs = buf ? &Bsh[1][0] : &Bsh[0][0];
#pragma unroll
    for (int j = 0; j < 4; ++j) {
      const bf16* q = bs + (j * 16 + (lane & 15)) * LROW + aoff;
      union { v16bf v; v8bf h[2]; } bu;
      bu.h[0] = *(const v8bf*)(q);
      bu.h[1] = *(const v8bf*)(q + 16);
      acc[j] = WMMA_BF16(a, bu.v, acc[j]);
    }

    __builtin_amdgcn_s_wait_asynccnt(0);       // my async write landed
    __syncthreads();                           // everyone's landed; reads of old buf done
    buf ^= 1;
  }
#else
  // ---------------- register-staged fallback pipeline ----------------
  v8bf bReg = *(const v8bf*)(gB);
  v8bf aLo  = *(const v8bf*)(gA);
  v8bf aHi  = *(const v8bf*)(gA + 16);

  for (int kc = 0; kc < nchunks; ++kc) {
    *(v8bf*)(buf ? sB1 : sB0) = bReg;
    union { v16bf v; v8bf h[2]; } au;
    au.h[0] = aLo; au.h[1] = aHi;
    v16bf a = au.v;

    int kn = (kc + 1) << 5;
    if (kn < K) {
      bReg = *(const v8bf*)(gB + kn);
      aLo  = *(const v8bf*)(gA + kn);
      aHi  = *(const v8bf*)(gA + kn + 16);
    }

    __syncthreads();

    const bf16* bs = buf ? &Bsh[1][0] : &Bsh[0][0];
#pragma unroll
    for (int j = 0; j < 4; ++j) {
      const bf16* q = bs + (j * 16 + (lane & 15)) * LROW + aoff;
      union { v16bf v; v8bf h[2]; } bu;
      bu.h[0] = *(const v8bf*)(q);
      bu.h[1] = *(const v8bf*)(q + 16);
      acc[j] = WMMA_BF16(a, bu.v, acc[j]);
    }
    buf ^= 1;
  }
#endif

  if (mBase < M) {                      // M % 16 == 0: tile fully valid or fully out
    int col  = lane & 15;
    int mOff = (lane >> 4) * 8;
#pragma unroll
    for (int j = 0; j < 4; ++j) {
      int n   = nBase + j * 16 + col;
      float b = bias[n];
#pragma unroll
      for (int i = 0; i < 8; ++i)
        C[(size_t)(mBase + mOff + i) * N + n] = acc[j][i] + b;
    }
  }
}

// ---------------- fp32 -> bf16 weight conversion ----------------
__global__ void cvt_f32_bf16(const float* __restrict__ src, bf16* __restrict__ dst, int n) {
  int i = blockIdx.x * blockDim.x + threadIdx.x;
  if (i < n) dst[i] = (bf16)src[i];
}

// ---------------- embedding gather + ReLU -> bf16 X[2016,512] ----
__global__ void embed_relu(const int* __restrict__ tgt, const float* __restrict__ emb,
                           bf16* __restrict__ X) {
  int idx = blockIdx.x * blockDim.x + threadIdx.x;   // NROWS*EMB
  if (idx >= NROWS * EMB) return;
  int row = idx >> 9;          // /EMB
  int e   = idx & (EMB - 1);
  int t   = row >> 5;          // /BATCH
  int b   = row & (BATCH - 1);
  int tok = (t == 0) ? SOS_TOK : tgt[b * MLEN + t];
  float x = emb[(size_t)tok * EMB + e];
  X[idx] = (bf16)(x > 0.f ? x : 0.f);
}

// ---------------- recurrent state init from encoder_hidden[2,B,H] -
__global__ void init_state(const float* __restrict__ eh,
                           float* __restrict__ h0f, bf16* __restrict__ h0b,
                           float* __restrict__ h1f, bf16* __restrict__ h1b) {
  int i = blockIdx.x * blockDim.x + threadIdx.x;     // BATCH*HID
  if (i >= BATCH * HID) return;
  float v0 = eh[i];
  float v1 = eh[BATCH * HID + i];
  h0f[i] = v0; h0b[i] = (bf16)v0;
  h1f[i] = v1; h1b[i] = (bf16)v1;
}

// ---------------- GRU gate combine (fp32 math) ------------------
__global__ void gru_combine(const float* __restrict__ gi,   // [B, 3H] (pre-offset)
                            const float* __restrict__ gh,   // [B, 3H]
                            float* __restrict__ hf,         // [B, H] in/out
                            bf16*  __restrict__ hbf,        // [B, H] out
                            bf16*  __restrict__ h1out) {    // H1 + t*HID or nullptr
  int idx = blockIdx.x * blockDim.x + threadIdx.x;   // BATCH*HID
  if (idx >= BATCH * HID) return;
  int b = idx >> 10;
  int j = idx & (HID - 1);
  const float* gir = gi + (size_t)b * G3;
  const float* ghr = gh + (size_t)b * G3;
  float r  = gir[j]           + ghr[j];
  float z  = gir[j + HID]     + ghr[j + HID];
  float hn = ghr[j + 2 * HID];
  r = 1.f / (1.f + expf(-r));
  z = 1.f / (1.f + expf(-z));
  float n = tanhf(gir[j + 2 * HID] + r * hn);
  float hnew = (1.f - z) * n + z * hf[idx];
  hf[idx]  = hnew;
  hbf[idx] = (bf16)hnew;
  if (h1out) h1out[(size_t)b * (TSTEP * HID) + j] = (bf16)hnew;
}

// ---------------------------------------------------------------
extern "C" void kernel_launch(void* const* d_in, const int* in_sizes, int n_in,
                              void* d_out, int out_size, void* d_ws, size_t ws_size,
                              hipStream_t stream) {
  (void)in_sizes; (void)n_in; (void)out_size; (void)ws_size;

  const float* enc_hidden = (const float*)d_in[1];   // [2,B,H]
  const int*   tgt        = (const int*)  d_in[2];   // [B,ML] int32
  const float* emb        = (const float*)d_in[3];   // [V,E]
  const float* W_ih0      = (const float*)d_in[4];   // [3H,E]
  const float* W_hh0      = (const float*)d_in[5];   // [3H,H]
  const float* b_ih0      = (const float*)d_in[6];
  const float* b_hh0      = (const float*)d_in[7];
  const float* W_ih1      = (const float*)d_in[8];   // [3H,H]
  const float* W_hh1      = (const float*)d_in[9];   // [3H,H]
  const float* b_ih1      = (const float*)d_in[10];
  const float* b_hh1      = (const float*)d_in[11];
  const float* W_out      = (const float*)d_in[12];  // [V,H]
  const float* b_out      = (const float*)d_in[13];
  float* out = (float*)d_out;                        // [B,T,V] fp32

  // -------- workspace carve-out (~120 MB) --------
  char* p = (char*)d_ws;
  auto carve = [&](size_t bytes) { char* r = p; p += (bytes + 255) & ~(size_t)255; return r; };
  bf16*  Wih0b = (bf16*)carve((size_t)G3 * EMB * 2);
  bf16*  Whh0b = (bf16*)carve((size_t)G3 * HID * 2);
  bf16*  Wih1b = (bf16*)carve((size_t)G3 * HID * 2);
  bf16*  Whh1b = (bf16*)carve((size_t)G3 * HID * 2);
  bf16*  Woutb = (bf16*)carve((size_t)VOC * HID * 2);
  bf16*  Xb    = (bf16*)carve((size_t)NROWS * EMB * 2);
  float* Gi0   = (float*)carve((size_t)NROWS * G3 * 4);
  float* gh0   = (float*)carve((size_t)BATCH * G3 * 4);
  float* gi1   = (float*)carve((size_t)BATCH * G3 * 4);
  float* gh1   = (float*)carve((size_t)BATCH * G3 * 4);
  float* h0f   = (float*)carve((size_t)BATCH * HID * 4);
  float* h1f   = (float*)carve((size_t)BATCH * HID * 4);
  bf16*  h0b   = (bf16*) carve((size_t)BATCH * HID * 2);
  bf16*  h1b   = (bf16*) carve((size_t)BATCH * HID * 2);
  bf16*  H1    = (bf16*) carve((size_t)NROWS * HID * 2);   // (b,t) row order

  auto cvt = [&](const float* s, bf16* d, int n) {
    cvt_f32_bf16<<<(n + 255) / 256, 256, 0, stream>>>(s, d, n);
  };
  auto gemmS = [&](const bf16* A, const bf16* W, const float* bias, float* C,
                   int M, int N, int K) {
    int waves  = (M / 16) * (N / 64);
    int blocks = (waves + 7) / 8;
    gemm_small<<<blocks, 256, 0, stream>>>(A, W, bias, C, M, N, K);
  };
  auto gemmB = [&](const bf16* A, const bf16* W, const float* bias, float* C,
                   int M, int N, int K) {
    dim3 g(N / BN, (M + BM - 1) / BM);
    gemm_big<<<g, 256, 0, stream>>>(A, W, bias, C, M, N, K);
  };

  // 1) weights -> bf16 (recomputed every call; L2-resident afterwards)
  cvt(W_ih0, Wih0b, G3 * EMB);
  cvt(W_hh0, Whh0b, G3 * HID);
  cvt(W_ih1, Wih1b, G3 * HID);
  cvt(W_hh1, Whh1b, G3 * HID);
  cvt(W_out, Woutb, VOC * HID);

  // 2) embeddings + state init + batched input-gate GEMM for layer 0
  embed_relu<<<(NROWS * EMB + 255) / 256, 256, 0, stream>>>(tgt, emb, Xb);
  init_state<<<(BATCH * HID + 255) / 256, 256, 0, stream>>>(enc_hidden, h0f, h0b, h1f, h1b);
  gemmB(Xb, Wih0b, b_ih0, Gi0, NROWS, G3, EMB);    // Gi0[t*B+b, 3H]

  // 3) sequential recurrence (63 steps)
  const int cblk = (BATCH * HID + 255) / 256;
  for (int t = 0; t < TSTEP; ++t) {
    gemmS(h0b, Whh0b, b_hh0, gh0, BATCH, G3, HID);
    gru_combine<<<cblk, 256, 0, stream>>>(Gi0 + (size_t)t * BATCH * G3, gh0,
                                          h0f, h0b, (bf16*)nullptr);
    gemmS(h0b, Wih1b, b_ih1, gi1, BATCH, G3, HID);
    gemmS(h1b, Whh1b, b_hh1, gh1, BATCH, G3, HID);
    gru_combine<<<cblk, 256, 0, stream>>>(gi1, gh1, h1f, h1b,
                                          H1 + (size_t)t * HID);
  }

  // 4) one big vocab projection: out[(b*T+t), :] = H1 @ W_out^T + b_out
  gemmB(H1, Woutb, b_out, out, NROWS, VOC, HID);
}
